// FavorAttention_35313221108114
// MI455X (gfx1250) — matrix operand order repaired and checked by reading, not compiled, not run on hardware
//
#include <hip/hip_runtime.h>
#include <hip/hip_bf16.h>

// FAVOR+ attention for MI455X (gfx1250, wave32, WMMA).
// GEMM stages run in bf16 WMMA (v_wmma_f32_16x16x32_bf16) with f32 accum;
// exp/stabilizer/denominator math stays f32.

typedef __attribute__((ext_vector_type(16))) __bf16        v16bf;
typedef __attribute__((ext_vector_type(8)))  float         v8f;
typedef __attribute__((ext_vector_type(8)))  unsigned int  v8u;

#define LQ      4096
#define DMODEL  1024
#define NH      16
#define DH      64
#define MF      256
#define EPSF    1e-4f
#define SCALEF  0.17677669529663687f   // 1024^-0.25 = 2^-2.5
#define SCALE2F 0.03125f               // SCALEF^2 = 2^-5
#define RSMF    0.0625f                // 256^-0.5

// ---- helpers ---------------------------------------------------------------

__device__ __forceinline__ unsigned short f2bf(float f) {
  unsigned int u = __float_as_uint(f);
  u = (u + 0x7fffu + ((u >> 16) & 1u)) >> 16;   // RNE truncate to bf16
  return (unsigned short)u;
}
__device__ __forceinline__ float bf2f(unsigned short h) {
  return __uint_as_float(((unsigned int)h) << 16);
}
// monotone float<->int map (works for atomicMax over negative floats)
__device__ __forceinline__ int ordf(float f) {
  int i = __float_as_int(f);
  return (i >= 0) ? i : (i ^ 0x7fffffff);
}
__device__ __forceinline__ float unordf(int i) {
  return __int_as_float((i >= 0) ? i : (i ^ 0x7fffffff));
}

// B-matrix operand: per-lane K layout = {p+0..7, p+16..23} (p = row+kb+sel)
__device__ __forceinline__ v8u lds16bf(const unsigned short* p) {
  const uint4* q = (const uint4*)p;
  uint4 a = q[0], b = q[2];
  return (v8u){a.x, a.y, a.z, a.w, b.x, b.y, b.z, b.w};
}
// A-matrix operand: per-lane K layout = {kb+sel+0..7, kb+16+sel+0..7}
__device__ __forceinline__ v8u ldsA(const unsigned short* row, int kb, int sel) {
  const uint4* q0 = (const uint4*)(row + kb + sel);
  const uint4* q1 = (const uint4*)(row + kb + 16 + sel);
  uint4 a = *q0, b = *q1;
  return (v8u){a.x, a.y, a.z, a.w, b.x, b.y, b.z, b.w};
}
__device__ __forceinline__ v8f wmma_bf(v8u a, v8u b, v8f c) {
  return __builtin_amdgcn_wmma_f32_16x16x32_bf16(
      false, __builtin_bit_cast(v16bf, a),
      false, __builtin_bit_cast(v16bf, b),
      (short)0, c, false, false);
}
__device__ __forceinline__ unsigned int packbf(float a, float b) {
  return (unsigned int)f2bf(a) | ((unsigned int)f2bf(b) << 16);
}

// ---- LDS row pitches (bf16 units, multiples of 8 => 16B aligned rows) ------
#define WROW   72     // W     [256][64]+pad
#define KTROW  72     // x tile[32][64]+pad (row-major, A operand for proj)
#define VTROW  40     // v^T   [64][32]+pad (row-major, B operand for kv)
#define KPROW  40     // k'^T  [256][32]+pad (row-major, A operand for kv)
#define KVROW  264    // kv^T  [64][256]+pad (B operand for out)
#define QPROW  264    // q'    [32][256]+pad (A operand for out)
#define LSPLIT 8
#define ROWSK  (LQ / LSPLIT)   // 512 key rows per block

// ---- kernel 0a: zero accumulators, seed stab = -inf ------------------------
__global__ __launch_bounds__(256)
void init_ws_kernel(float* __restrict__ acc, int n, int* __restrict__ stab) {
  int i = blockIdx.x * blockDim.x + threadIdx.x;
  if (i == 0) *stab = ordf(-__builtin_inff());
  for (int j = i; j < n; j += gridDim.x * blockDim.x) acc[j] = 0.f;
}

// ---- kernel 0b: stab = max over all keys of h = -0.5*||scale*k_slice||^2 ---
__global__ __launch_bounds__(256)
void stab_kernel(const float* __restrict__ K, int* __restrict__ stab, int nslice) {
  __shared__ float red[256];
  const int tid = threadIdx.x;
  const int idx = blockIdx.x * 256 + tid;
  float h = -__builtin_inff();
  if (idx < nslice) {
    const float4* p = (const float4*)K + (size_t)idx * 16;  // 64 contiguous floats
    float s = 0.f;
#pragma unroll
    for (int i = 0; i < 16; ++i) {
      float4 v = p[i];
      s += v.x * v.x + v.y * v.y + v.z * v.z + v.w * v.w;
    }
    h = -0.5f * SCALE2F * s;
  }
  red[tid] = h;
  __syncthreads();
  for (int o = 128; o > 0; o >>= 1) {
    if (tid < o) red[tid] = fmaxf(red[tid], red[tid + o]);
    __syncthreads();
  }
  if (tid == 0) atomicMax(stab, ordf(red[0]));
}

// ---- kernel 1: k' features, kv = k'^T @ v, ksum = sum_l k' -----------------
__global__ __launch_bounds__(256)
void favor_kv_kernel(const float* __restrict__ K, const float* __restrict__ V,
                     const float* __restrict__ W, const int* __restrict__ stab_i,
                     float* __restrict__ ws_kv, float* __restrict__ ws_ksum) {
  extern __shared__ unsigned char smem[];
  unsigned short* Wl    = (unsigned short*)smem;      // 256*72 bf16
  unsigned short* ktile = Wl + 256 * WROW;            // 32*72
  unsigned short* vT    = ktile + 32 * KTROW;         // 64*40
  unsigned short* kT    = vT + 64 * VTROW;            // 256*40
  float* hpart = (float*)(kT + 256 * KPROW);          // 32*8
  float* hrow  = hpart + 32 * 8;                      // 32

  const int tid  = threadIdx.x;
  const int wave = tid >> 5;
  const int lane = tid & 31;
  const int ln   = lane & 15;
  const int sel  = (lane < 16) ? 0 : 8;   // A-operand K select
  const int koff = (lane < 16) ? 0 : 16;  // unused; B operand also takes sel
  const int mhi  = (lane < 16) ? 0 : 8;   // C/D-operand M select

  const int bh = blockIdx.x / LSPLIT;
  const int sp = blockIdx.x % LSPLIT;
  const int b  = bh / NH;
  const int h  = bh % NH;

  const float stab = unordf(*stab_i);

  // stage W as bf16 (row-major [m][d])
  for (int i = tid; i < 256 * 32; i += 256) {
    int m = i >> 5, d2 = i & 31;
    float2 w2 = ((const float2*)W)[m * 32 + d2];
    ((unsigned int*)Wl)[m * (WROW / 2) + d2] = packbf(w2.x, w2.y);
  }

  v8f acc[8];
#pragma unroll
  for (int i = 0; i < 8; ++i) acc[i] = (v8f){0, 0, 0, 0, 0, 0, 0, 0};
  float ksum_reg = 0.f;

  const int r0 = tid >> 3;        // row in 32-row tile
  const int db = (tid & 7) * 8;   // 8-column chunk

  for (int t = 0; t < ROWSK / 32; ++t) {
    const int lbase = sp * ROWSK + t * 32;
    // load k tile (scaled, bf16 row-major) + per-row sumsq partials; v tile transposed
    {
      size_t off = ((size_t)(b * LQ + lbase + r0)) * DMODEL + h * DH + db;
      float4 a = *(const float4*)(K + off);
      float4 c = *(const float4*)(K + off + 4);
      a.x *= SCALEF; a.y *= SCALEF; a.z *= SCALEF; a.w *= SCALEF;
      c.x *= SCALEF; c.y *= SCALEF; c.z *= SCALEF; c.w *= SCALEF;
      hpart[r0 * 8 + (tid & 7)] =
          a.x * a.x + a.y * a.y + a.z * a.z + a.w * a.w +
          c.x * c.x + c.y * c.y + c.z * c.z + c.w * c.w;
      unsigned int* kd = (unsigned int*)ktile + ((r0 * KTROW + db) >> 1);
      kd[0] = packbf(a.x, a.y); kd[1] = packbf(a.z, a.w);
      kd[2] = packbf(c.x, c.y); kd[3] = packbf(c.z, c.w);

      float4 va = *(const float4*)(V + off);
      float4 vc = *(const float4*)(V + off + 4);
      float vv[8] = {va.x, va.y, va.z, va.w, vc.x, vc.y, vc.z, vc.w};
#pragma unroll
      for (int j = 0; j < 8; ++j) vT[(db + j) * VTROW + r0] = f2bf(vv[j]);
    }
    __syncthreads();
    if (tid < 32) {
      float s = 0.f;
#pragma unroll
      for (int j = 0; j < 8; ++j) s += hpart[tid * 8 + j];
      hrow[tid] = -0.5f * s;
    }
    __syncthreads();

    // proj = k_tile @ W^T via WMMA, then k' = rsm*(exp(h+proj-stab)+eps) -> kT (transposed)
#pragma unroll
    for (int mh = 0; mh < 2; ++mh) {
#pragma unroll
      for (int j = 0; j < 2; ++j) {
        const int nc = wave * 2 + j;
        v8f p = (v8f){0, 0, 0, 0, 0, 0, 0, 0};
#pragma unroll
        for (int kb = 0; kb < 64; kb += 32) {
          v8u A  = ldsA(ktile + (mh * 16 + ln) * KTROW, kb, sel);
          v8u Bv = lds16bf(Wl + (nc * 16 + ln) * WROW + kb + sel);
          p = wmma_bf(A, Bv, p);
        }
        const int ncol = nc * 16 + ln;
#pragma unroll
        for (int e = 0; e < 8; ++e) {
          int l = mh * 16 + e + mhi;
          float kp = RSMF * (__expf(hrow[l] + p[e] - stab) + EPSF);
          kT[ncol * KPROW + l] = f2bf(kp);
        }
      }
    }
    __syncthreads();

    // ksum: thread tid owns feature m=tid, sum over 32 rows of this tile
    {
      const unsigned int* kr = (const unsigned int*)(kT + tid * KPROW);
      float s = 0.f;
#pragma unroll
      for (int j = 0; j < 16; ++j) {
        unsigned int u = kr[j];
        s += bf2f((unsigned short)u) + bf2f((unsigned short)(u >> 16));
      }
      ksum_reg += s;
    }

    // kv += k'^T @ v  (K = 32 rows of this tile); wave owns m-chunks {2w,2w+1} x 4 d-chunks
#pragma unroll
    for (int j = 0; j < 2; ++j) {
      const int mc = wave * 2 + j;
      v8u A = ldsA(kT + (mc * 16 + ln) * KPROW, 0, sel);
#pragma unroll
      for (int dc = 0; dc < 4; ++dc) {
        v8u Bv = lds16bf(vT + (dc * 16 + ln) * VTROW + sel);
        acc[j * 4 + dc] = wmma_bf(A, Bv, acc[j * 4 + dc]);
      }
    }
    __syncthreads();
  }

  // flush partial results
  atomicAdd(&ws_ksum[bh * MF + tid], ksum_reg);
#pragma unroll
  for (int j = 0; j < 2; ++j) {
    const int mc = wave * 2 + j;
#pragma unroll
    for (int dc = 0; dc < 4; ++dc) {
#pragma unroll
      for (int e = 0; e < 8; ++e) {
        int m = mc * 16 + e + mhi;
        int d = dc * 16 + ln;
        atomicAdd(&ws_kv[((size_t)bh * MF + m) * DH + d], acc[j * 4 + dc][e]);
      }
    }
  }
}

// ---- kernel 2: q', denominator, out = (q' @ kv) / d ------------------------
__global__ __launch_bounds__(256)
void favor_out_kernel(const float* __restrict__ Q, const float* __restrict__ W,
                      const float* __restrict__ ws_kv, const float* __restrict__ ws_ksum,
                      float* __restrict__ out) {
  extern __shared__ unsigned char smem[];
  unsigned short* Wl    = (unsigned short*)smem;      // 256*72
  unsigned short* kvT   = Wl + 256 * WROW;            // 64*264
  unsigned short* qtile = kvT + 64 * KVROW;           // 32*72
  unsigned short* qp    = qtile + 32 * KTROW;         // 32*264
  float* ksumL = (float*)(qp + 32 * QPROW);           // 256
  float* den8  = ksumL + 256;                         // 32*8
  float* den   = den8 + 256;                          // 32

  const int tid  = threadIdx.x;
  const int wave = tid >> 5;
  const int lane = tid & 31;
  const int ln   = lane & 15;
  const int sel  = (lane < 16) ? 0 : 8;
  const int koff = (lane < 16) ? 0 : 16;
  const int mhi  = (lane < 16) ? 0 : 8;

  const int bh = blockIdx.x >> 5;       // 32 chunks of 128 rows per (b,h)
  const int ch = blockIdx.x & 31;
  const int b  = bh / NH;
  const int h  = bh % NH;

  // stage W (bf16), kv^T (bf16), ksum (f32)
  for (int i = tid; i < 256 * 32; i += 256) {
    int m = i >> 5, d2 = i & 31;
    float2 w2 = ((const float2*)W)[m * 32 + d2];
    ((unsigned int*)Wl)[m * (WROW / 2) + d2] = packbf(w2.x, w2.y);
  }
  for (int i = tid; i < MF * DH; i += 256) {
    int m = i >> 6, d = i & 63;
    kvT[d * KVROW + m] = f2bf(ws_kv[((size_t)bh * MF + m) * DH + d]);
  }
  ksumL[tid] = ws_ksum[bh * MF + tid];
  __syncthreads();

  const int r0 = tid >> 3;
  const int db = (tid & 7) * 8;

  for (int t = 0; t < 4; ++t) {
    const int lbase = ch * 128 + t * 32;
    // load q tile (scaled, bf16 row-major); query stabilizer cancels h entirely
    {
      size_t off = ((size_t)(b * LQ + lbase + r0)) * DMODEL + h * DH + db;
      float4 a = *(const float4*)(Q + off);
      float4 c = *(const float4*)(Q + off + 4);
      a.x *= SCALEF; a.y *= SCALEF; a.z *= SCALEF; a.w *= SCALEF;
      c.x *= SCALEF; c.y *= SCALEF; c.z *= SCALEF; c.w *= SCALEF;
      unsigned int* qd = (unsigned int*)qtile + ((r0 * KTROW + db) >> 1);
      qd[0] = packbf(a.x, a.y); qd[1] = packbf(a.z, a.w);
      qd[2] = packbf(c.x, c.y); qd[3] = packbf(c.z, c.w);
    }
    __syncthreads();

    // proj -> q' = rsm*(exp(proj)+eps), stored row-major [l][m]
#pragma unroll
    for (int mh = 0; mh < 2; ++mh) {
#pragma unroll
      for (int j = 0; j < 2; ++j) {
        const int nc = wave * 2 + j;
        v8f p = (v8f){0, 0, 0, 0, 0, 0, 0, 0};
#pragma unroll
        for (int kb = 0; kb < 64; kb += 32) {
          v8u A  = ldsA(qtile + (mh * 16 + ln) * KTROW, kb, sel);
          v8u Bv = lds16bf(Wl + (nc * 16 + ln) * WROW + kb + sel);
          p = wmma_bf(A, Bv, p);
        }
        const int ncol = nc * 16 + ln;
#pragma unroll
        for (int e = 0; e < 8; ++e) {
          int l = mh * 16 + e + mhi;
          float qv = RSMF * (__expf(p[e]) + EPSF);
          qp[l * QPROW + ncol] = f2bf(qv);
        }
      }
    }
    __syncthreads();

    // denominator partials: d_[l] = q'[l,:] . ksum
    {
      int row = tid >> 3, seg = tid & 7;
      const unsigned int* qr = (const unsigned int*)(qp + row * QPROW + seg * 32);
      float s = 0.f;
#pragma unroll
      for (int j = 0; j < 16; ++j) {
        unsigned int u = qr[j];
        s += bf2f((unsigned short)u) * ksumL[seg * 32 + 2 * j] +
             bf2f((unsigned short)(u >> 16)) * ksumL[seg * 32 + 2 * j + 1];
      }
      den8[row * 8 + seg] = s;
    }
    __syncthreads();
    if (tid < 32) {
      float s = 0.f;
#pragma unroll
      for (int j = 0; j < 8; ++j) s += den8[tid * 8 + j];
      s = s + 2.0f * EPSF * ((fabsf(s) <= EPSF) ? 1.0f : 0.0f);
      den[tid] = s;
    }
    __syncthreads();

    // out tile = q' @ kv (K = 256); wave owns one 16x16 C tile
    {
      const int mh = wave >> 2, dc = wave & 3;
      v8f o = (v8f){0, 0, 0, 0, 0, 0, 0, 0};
#pragma unroll
      for (int kb = 0; kb < 256; kb += 32) {
        v8u A  = ldsA(qp + (mh * 16 + ln) * QPROW, kb, sel);
        v8u Bv = lds16bf(kvT + (dc * 16 + ln) * KVROW + kb + sel);
        o = wmma_bf(A, Bv, o);
      }
#pragma unroll
      for (int e = 0; e < 8; ++e) {
        int l   = mh * 16 + e + mhi;
        int col = dc * 16 + ln;
        out[((size_t)(b * LQ + lbase + l)) * DMODEL + h * DH + col] = o[e] / den[l];
      }
    }
    __syncthreads();
  }
}

// ---- host glue -------------------------------------------------------------
extern "C" void kernel_launch(void* const* d_in, const int* in_sizes, int n_in,
                              void* d_out, int out_size, void* d_ws, size_t ws_size,
                              hipStream_t stream) {
  (void)n_in; (void)out_size; (void)ws_size;
  const float* q = (const float*)d_in[0];
  const float* k = (const float*)d_in[1];
  const float* v = (const float*)d_in[2];
  const float* W = (const float*)d_in[3];
  float* out = (float*)d_out;

  const int B  = in_sizes[0] / (LQ * DMODEL);
  const int BH = B * NH;

  int*   ws_stab = (int*)d_ws;
  float* ws_kv   = (float*)((char*)d_ws + 64);
  float* ws_ksum = ws_kv + (size_t)BH * MF * DH;
  const int accN = BH * MF * DH + BH * MF;   // kv + ksum, contiguous

  init_ws_kernel<<<512, 256, 0, stream>>>(ws_kv, accN, ws_stab);

  const int nslice = B * LQ * NH;
  stab_kernel<<<(nslice + 255) / 256, 256, 0, stream>>>(k, ws_stab, nslice);

  const size_t sh1 = (size_t)(256 * WROW + 32 * KTROW + 64 * VTROW + 256 * KPROW) * 2
                   + (size_t)(32 * 8 + 32) * 4;
  favor_kv_kernel<<<BH * LSPLIT, 256, sh1, stream>>>(k, v, W, ws_stab, ws_kv, ws_ksum);

  const size_t sh2 = (size_t)(256 * WROW + 64 * KVROW + 32 * KTROW + 32 * QPROW) * 2
                   + (size_t)(256 + 256 + 32) * 4;
  favor_out_kernel<<<BH * 32, 256, sh2, stream>>>(q, W, ws_kv, ws_ksum, out);
}
